// EqV3StyleNodeFFN_63986422776406
// MI455X (gfx1250) — matrix operand order repaired and checked by reading, not compiled, not run on hardware
//
#include <hip/hip_runtime.h>
#include <hip/hip_bf16.h>

typedef __bf16 bf16_t;
typedef __attribute__((ext_vector_type(16))) __bf16 v16bf;
typedef __attribute__((ext_vector_type(8)))  __bf16 v8bf;
typedef __attribute__((ext_vector_type(8)))  float  v8f;

#define NT        16      // nodes per tile (WMMA M dim)
#define MCOEFF    9       // spherical coefficients
#define CCH       128
#define HCH       128
#define AGRID     42
#define FLAT_DIM  1152    // M*C

__device__ __forceinline__ int deg_of(int m)  { return (m == 0) ? 0 : ((m < 4) ? 1 : 2); }
__device__ __forceinline__ int deg_off(int l) { return (l == 0) ? 0 : ((l == 1) ? 128 : 512); }

__device__ __forceinline__ v8f wmma_bf16(v16bf a, v16bf b, v8f c) {
  return __builtin_amdgcn_wmma_f32_16x16x32_bf16(false, a, false, b, (short)0, c, false, false);
}

// CDNA5 async DMA: global -> LDS, 16B per lane, tracked by ASYNCcnt.
__device__ __forceinline__ void async_load_b128(unsigned lds_off, const void* gaddr) {
  asm volatile("global_load_async_to_lds_b128 %0, %1, off"
               :: "v"(lds_off), "v"(gaddr) : "memory");
}
__device__ __forceinline__ void wait_asynccnt0() {
  asm volatile("s_wait_asynccnt 0x0" ::: "memory");
}

// A fragment: 16 rows x 32 k (bf16), base already offset to k-block start,
// row-major, stride 128 bf16. Lane<16: K 0..15; lane>=16: K 16..31.
__device__ __forceinline__ v16bf load_a(const bf16_t* base, int lane) {
  const int row = lane & 15;
  const int ks  = (lane >> 4) * 16;
  const bf16_t* p = base + row * 128 + ks;
  v8bf lo = *(const v8bf*)(p);
  v8bf hi = *(const v8bf*)(p + 8);
  v16bf f;
#pragma unroll
  for (int i = 0; i < 8; ++i) { f[i] = lo[i]; f[i + 8] = hi[i]; }
  return f;
}

// B fragment from global fp32 weights laid out [col][K] (K contiguous):
// p points at this lane's 16 contiguous floats; convert to bf16.
__device__ __forceinline__ v16bf load_b_gf32(const float* p) {
  v16bf f;
#pragma unroll
  for (int i = 0; i < 16; i += 4) {
    float4 v = *(const float4*)(p + i);
    f[i] = (bf16_t)v.x; f[i + 1] = (bf16_t)v.y;
    f[i + 2] = (bf16_t)v.z; f[i + 3] = (bf16_t)v.w;
  }
  return f;
}

__global__ __launch_bounds__(256, 1)
void EqV3StyleNodeFFN_63986422776406_kernel(
    const float* __restrict__ x,    const float* __restrict__ nw,
    const float* __restrict__ w1,   const float* __restrict__ b1,
    const float* __restrict__ gw,   const float* __restrict__ gb,
    const float* __restrict__ w2,   const float* __restrict__ b2,
    const float* __restrict__ tg,   const float* __restrict__ fg,
    float* __restrict__ out, int Ntot)
{
  extern __shared__ char smem_raw[];
  // x staging region (73,728 B) is dead after the pack stage; h2_bf aliases it.
  float*  x_lds = (float*)smem_raw;                        // [16][1152] fp32
  bf16_t* h2_bf = (bf16_t*)smem_raw;                       // [9][16][128] (alias)
  bf16_t* p_bf  = (bf16_t*)(smem_raw + 73728);             // [9][16][128]
  bf16_t* h1_bf = p_bf  + MCOEFF * NT * HCH;               // [9][16][128]
  bf16_t* g_a   = h1_bf + MCOEFF * NT * HCH;               // [16][128]
  bf16_t* g2_a  = g_a   + NT * HCH;                        // [16][128]
  float*  red   = (float*)(g2_a + NT * HCH);               // [16][17]
  float*  rinv  = red + 16 * 17;                           // [16]

  const int tid  = threadIdx.x;
  const int wave = tid >> 5;
  const int lane = tid & 31;
  const int nodeBase = blockIdx.x * NT;

  __builtin_prefetch(w1, 0, 1);
  __builtin_prefetch(w2, 0, 1);

  // ---- stage 0: async-DMA the 16-node x tile into LDS (single HBM pass over x)
  {
    const int rowVec = FLAT_DIM / 4;              // 288 float4 per node
    const int nvec   = NT * rowVec;               // 4608 float4
    for (int i = tid; i < nvec; i += 256) {
      const int n = i / rowVec;
      const int j = i - n * rowVec;
      int gn = nodeBase + n; if (gn >= Ntot) gn = Ntot - 1;
      async_load_b128((unsigned)(size_t)(x_lds + (size_t)i * 4),
                      x + (size_t)gn * FLAT_DIM + (size_t)j * 4);
    }
    wait_asynccnt0();
  }
  __syncthreads();

  // ---- stage 1: per-node sum of squares (16 threads per node, from LDS)
  {
    const int n = tid >> 4, j = tid & 15;
    const float* xr = x_lds + n * FLAT_DIM;
    float s = 0.f;
    for (int k = j; k < FLAT_DIM; k += 16) { float v = xr[k]; s += v * v; }
    red[n * 17 + j] = s;
  }
  __syncthreads();
  if (tid < 16) {
    float s = 0.f;
#pragma unroll
    for (int j = 0; j < 16; ++j) s += red[tid * 17 + j];
    rinv[tid] = rsqrtf(s * (1.0f / (float)FLAT_DIM) + 1e-6f);
  }
  __syncthreads();

  // ---- stage 1b: pack + normalize + affine -> p_bf[m][n][c]  (gather from LDS)
  for (int e = tid; e < MCOEFF * NT * CCH; e += 256) {
    const int c = e & 127, n = (e >> 7) & 15, m = e >> 11;
    const int l = deg_of(m), dim = 2 * l + 1, mloc = m - l * l;
    const float v = x_lds[n * FLAT_DIM + deg_off(l) + c * dim + mloc];
    p_bf[e] = (bf16_t)(v * nw[l * CCH + c] * rinv[n]);
  }
  __syncthreads();

  // ---- stage 2: per-coeff GEMM  h1[n,h] = p[n,:] @ w1[l]^T   (WMMA, waves own coeffs)
  for (int m = wave; m < MCOEFF; m += 8) {
    const int l = deg_of(m);
    const bf16_t* Abase = p_bf + m * NT * CCH;
#pragma unroll
    for (int ct = 0; ct < 8; ++ct) {
      v8f acc = {0.f, 0.f, 0.f, 0.f, 0.f, 0.f, 0.f, 0.f};
      const int h = ct * 16 + (lane & 15);
#pragma unroll
      for (int kb = 0; kb < 4; ++kb) {
        v16bf a = load_a(Abase + kb * 32, lane);
        v16bf b = load_b_gf32(w1 + ((size_t)(l * HCH + h)) * CCH + kb * 32 + (lane >> 4) * 16);
        acc = wmma_bf16(a, b, acc);
      }
      const int roff = (lane >> 4) * 8;
      const float bias = (m == 0) ? b1[h] : 0.f;
#pragma unroll
      for (int r = 0; r < 8; ++r)
        h1_bf[(m * NT + (r + roff)) * HCH + h] = (bf16_t)(acc[r] + bias);
    }
  }

  // ---- hoist loop-invariant SwiGLU B fragments (grid_w) into registers:
  // wave w owns output column pair (tile w -> gate half, tile w+8 -> value half)
  v16bf gwA[4], gwB[4];
  {
    const int colA = wave * 16 + (lane & 15);
    const int colB = 128 + colA;
#pragma unroll
    for (int kb = 0; kb < 4; ++kb) {
      gwA[kb] = load_b_gf32(gw + (size_t)colA * HCH + kb * 32 + (lane >> 4) * 16);
      gwB[kb] = load_b_gf32(gw + (size_t)colB * HCH + kb * 32 + (lane >> 4) * 16);
    }
  }
  __syncthreads();

  // ---- per-thread ownership for the VALU contractions
  const int own_n  = tid & 15;
  const int own_h0 = (tid >> 4) * 8;
  float h2r[MCOEFF][8];
#pragma unroll
  for (int m = 0; m < MCOEFF; ++m)
#pragma unroll
    for (int q = 0; q < 8; ++q) h2r[m][q] = 0.f;

  // ---- grid-point loop: to_grid (VALU) -> SwiGLU (WMMA) -> from_grid (VALU accum)
  for (int a = 0; a < AGRID; ++a) {
    // to_grid: g_a[n][h] = sum_m T[a,m] * h1[n,m,h]
    {
      float gacc[8] = {0.f, 0.f, 0.f, 0.f, 0.f, 0.f, 0.f, 0.f};
#pragma unroll
      for (int m = 0; m < MCOEFF; ++m) {
        const float t = tg[a * MCOEFF + m];
        v8bf hv = *(const v8bf*)(h1_bf + (m * NT + own_n) * HCH + own_h0);
#pragma unroll
        for (int q = 0; q < 8; ++q) gacc[q] += t * (float)hv[q];
      }
#pragma unroll
      for (int q = 0; q < 8; ++q) g_a[own_n * HCH + own_h0 + q] = (bf16_t)gacc[q];
    }
    __syncthreads();

    // SwiGLU: z = g_a @ grid_w^T + grid_b ; g2 = silu(z[:,:H]) * z[:,H:]
    {
      v8f zA = {0.f, 0.f, 0.f, 0.f, 0.f, 0.f, 0.f, 0.f};
      v8f zB = {0.f, 0.f, 0.f, 0.f, 0.f, 0.f, 0.f, 0.f};
#pragma unroll
      for (int kb = 0; kb < 4; ++kb) {
        v16bf afr = load_a(g_a + kb * 32, lane);
        zA = wmma_bf16(afr, gwA[kb], zA);
        zB = wmma_bf16(afr, gwB[kb], zB);
      }
      const int colh = wave * 16 + (lane & 15);
      const int roff = (lane >> 4) * 8;
      const float bA = gb[colh], bB = gb[128 + colh];
#pragma unroll
      for (int r = 0; r < 8; ++r) {
        const float za = zA[r] + bA;
        const float zb = zB[r] + bB;
        const float s  = za / (1.f + __expf(-za));   // silu
        g2_a[(r + roff) * HCH + colh] = (bf16_t)(s * zb);
      }
    }
    __syncthreads();

    // from_grid: h2[n,m,h] += F[m,a] * g2[n,h]
    {
      float gv[8];
      v8bf g2v = *(const v8bf*)(g2_a + own_n * HCH + own_h0);
#pragma unroll
      for (int q = 0; q < 8; ++q) gv[q] = (float)g2v[q];
#pragma unroll
      for (int m = 0; m < MCOEFF; ++m) {
        const float f = fg[m * AGRID + a];
#pragma unroll
        for (int q = 0; q < 8; ++q) h2r[m][q] += f * gv[q];
      }
    }
    // g_a rewrite (next iter) is fenced by the barrier after SwiGLU;
    // g2_a rewrite is fenced by the next to_grid barrier -> two barriers/iter suffice.
  }

  // ---- spill h2 accumulators to LDS (bf16, aliases dead x staging region)
#pragma unroll
  for (int m = 0; m < MCOEFF; ++m)
#pragma unroll
    for (int q = 0; q < 8; ++q)
      h2_bf[(m * NT + own_n) * HCH + own_h0 + q] = (bf16_t)h2r[m][q];
  __syncthreads();

  // ---- stage 6: per-coeff GEMM  out[n,c] = h2[n,:] @ w2[l]^T, fused unpack-scatter
  for (int m = wave; m < MCOEFF; m += 8) {
    const int l = deg_of(m), dim = 2 * l + 1, mloc = m - l * l, base = deg_off(l);
    const bf16_t* Abase = h2_bf + m * NT * HCH;
#pragma unroll
    for (int ct = 0; ct < 8; ++ct) {
      v8f acc = {0.f, 0.f, 0.f, 0.f, 0.f, 0.f, 0.f, 0.f};
      const int ccol = ct * 16 + (lane & 15);
#pragma unroll
      for (int kb = 0; kb < 4; ++kb) {
        v16bf a = load_a(Abase + kb * 32, lane);
        v16bf b = load_b_gf32(w2 + ((size_t)(l * CCH + ccol)) * HCH + kb * 32 + (lane >> 4) * 16);
        acc = wmma_bf16(a, b, acc);
      }
      const int roff = (lane >> 4) * 8;
      const float bias = (m == 0) ? b2[ccol] : 0.f;
#pragma unroll
      for (int r = 0; r < 8; ++r) {
        const int gn = nodeBase + r + roff;
        if (gn < Ntot)
          out[(size_t)gn * FLAT_DIM + base + ccol * dim + mloc] = acc[r] + bias;
      }
    }
  }
}

extern "C" void kernel_launch(void* const* d_in, const int* in_sizes, int n_in,
                              void* d_out, int out_size, void* d_ws, size_t ws_size,
                              hipStream_t stream) {
  const float* x  = (const float*)d_in[0];
  const float* nw = (const float*)d_in[1];
  const float* w1 = (const float*)d_in[2];
  const float* b1 = (const float*)d_in[3];
  const float* gw = (const float*)d_in[4];
  const float* gb = (const float*)d_in[5];
  const float* w2 = (const float*)d_in[6];
  const float* b2 = (const float*)d_in[7];
  const float* tg = (const float*)d_in[8];
  const float* fg = (const float*)d_in[9];
  float* out = (float*)d_out;

  const int Ntot   = in_sizes[0] / FLAT_DIM;
  const int blocks = (Ntot + NT - 1) / NT;

  const size_t smem = 73728                                          // x tile (aliased by h2)
      + (size_t)(2 * MCOEFF * NT * HCH + 2 * NT * HCH) * sizeof(bf16_t)  // p, h1, g_a, g2_a
      + (size_t)(16 * 17 + 16) * sizeof(float);                      // reductions

  hipFuncSetAttribute((const void*)EqV3StyleNodeFFN_63986422776406_kernel,
                      hipFuncAttributeMaxDynamicSharedMemorySize, (int)smem);

  EqV3StyleNodeFFN_63986422776406_kernel<<<blocks, 256, smem, stream>>>(
      x, nw, w1, b1, gw, gb, w2, b2, tg, fg, out, Ntot);
}
